// VSAETopK_21595095564421
// MI455X (gfx1250) — compile-verified, matrix operand order checked by reading, not run on hardware
//
#include <hip/hip_runtime.h>

#define BATCH  4096
#define D_ACT  768
#define D_DICT 24576
#define KSEL   64

typedef __attribute__((ext_vector_type(16))) __bf16 v16bf;
typedef __attribute__((ext_vector_type(8)))  __bf16 bf16x8;
typedef __attribute__((ext_vector_type(4)))  __bf16 bf16x4;
typedef __attribute__((ext_vector_type(8)))  float  v8f;

// ---------------------------------------------------------------------------
// fp32 -> bf16 conversion (vectorized by 4)
// ---------------------------------------------------------------------------
__global__ __launch_bounds__(256) void cvt_bf16_kernel(const float* __restrict__ in,
                                                       __bf16* __restrict__ out,
                                                       long n4) {
    long i = (long)blockIdx.x * blockDim.x + threadIdx.x;
    if (i >= n4) return;
    const float4 v = ((const float4*)in)[i];
    bf16x4 o = {(__bf16)v.x, (__bf16)v.y, (__bf16)v.z, (__bf16)v.w};
    ((bf16x4*)out)[i] = o;
}

// ---------------------------------------------------------------------------
// Encoder GEMM: mu[B, D_DICT] = relu(x[B, D_ACT] * W_enc[D_DICT, D_ACT]^T + b_enc)
// 256 threads = 8 waves (wave32): 2 waves over M x 4 waves over N.
// Block tile 128x256, wave tile 64x64 (4 A-frags x 4 B-frags -> 16 WMMA / K-step).
// K-step 32 (v_wmma_f32_16x16x32_bf16, f32 accumulate).
// mu is written with non-temporal stores (read once by topk, then dead;
// keeps the 192MB L2 holding W_enc instead).
// ---------------------------------------------------------------------------
__global__ __launch_bounds__(256) void enc_wmma_kernel(const __bf16* __restrict__ X,
                                                       const __bf16* __restrict__ W,
                                                       const float* __restrict__ b_enc,
                                                       float* __restrict__ mu) {
    const int lane = threadIdx.x & 31;
    const int wave = threadIdx.x >> 5;
    const int half = lane >> 4;   // K-group selector (16-bit A/B fragment layout)
    const int r16  = lane & 15;   // row (A) / col (B) within the 16-wide tile
    const int wm   = wave & 1;    // 0..1 along M
    const int wn   = wave >> 1;   // 0..3 along N
    const int MB   = blockIdx.x * 128 + wm * 64;
    const int NB   = blockIdx.y * 256 + wn * 64;

    v8f acc[4][4];
#pragma unroll
    for (int t = 0; t < 4; ++t)
#pragma unroll
        for (int j = 0; j < 4; ++j)
            acc[t][j] = (v8f){0.f, 0.f, 0.f, 0.f, 0.f, 0.f, 0.f, 0.f};

    for (int kk = 0; kk < D_ACT; kk += 32) {
        const int kfrag = kk + half * 8;

        v16bf afrag[4];
#pragma unroll
        for (int t = 0; t < 4; ++t) {
            const __bf16* pa = X + (size_t)(MB + 16 * t + r16) * D_ACT + kfrag;
            bf16x8 lo = *(const bf16x8*)pa;
            bf16x8 hi = *(const bf16x8*)(pa + 16);
            afrag[t] = __builtin_shufflevector(lo, hi, 0, 1, 2, 3, 4, 5, 6, 7,
                                               8, 9, 10, 11, 12, 13, 14, 15);
        }

        v16bf bfrag[4];
#pragma unroll
        for (int j = 0; j < 4; ++j) {
            const __bf16* pb = W + (size_t)(NB + 16 * j + r16) * D_ACT + kfrag;
            bf16x8 lo = *(const bf16x8*)pb;
            bf16x8 hi = *(const bf16x8*)(pb + 16);
            bfrag[j] = __builtin_shufflevector(lo, hi, 0, 1, 2, 3, 4, 5, 6, 7,
                                               8, 9, 10, 11, 12, 13, 14, 15);
            // pull the next K-tile of this W row toward the WGP
            __builtin_prefetch(pb + 32, 0, 3);
        }

#pragma unroll
        for (int t = 0; t < 4; ++t)
#pragma unroll
            for (int j = 0; j < 4; ++j)
                acc[t][j] = __builtin_amdgcn_wmma_f32_16x16x32_bf16(
                    /*neg_a=*/false, afrag[t], /*neg_b=*/false, bfrag[j],
                    /*c_mod=*/(short)0, acc[t][j],
                    /*reuse_a=*/false, /*reuse_b=*/false);
    }

    // Epilogue: + b_enc, relu, NT-store.  C layout: VGPR r -> row r+8*half, col r16.
#pragma unroll
    for (int t = 0; t < 4; ++t) {
#pragma unroll
        for (int j = 0; j < 4; ++j) {
            const int col = NB + 16 * j + r16;
            const float bias = b_enc[col];
#pragma unroll
            for (int r = 0; r < 8; ++r) {
                const int row = MB + 16 * t + 8 * half + r;
                float v = acc[t][j][r] + bias;
                v = v > 0.f ? v : 0.f;
                __builtin_nontemporal_store(v, &mu[(size_t)row * D_DICT + col]);
            }
        }
    }
}

// ---------------------------------------------------------------------------
// Top-K per row: stage full row (96 KB) in LDS via non-temporal loads, then
// 64 iterative argmax passes.  Reduction: wave32 shfl_xor + one LDS step.
// One workgroup (256 threads = 8 waves) per row.
// ---------------------------------------------------------------------------
__global__ __launch_bounds__(256) void topk_kernel(const float* __restrict__ mu,
                                                   float* __restrict__ vals,
                                                   int* __restrict__ idxs) {
    __shared__ float buf[D_DICT];    // 96 KB of the 320 KB WGP LDS
    __shared__ float wmaxs[8];
    __shared__ int   wlocs[8];

    const int row  = blockIdx.x;
    const int tid  = threadIdx.x;
    const int lane = tid & 31;
    const int wave = tid >> 5;
    const float* src = mu + (size_t)row * D_DICT;

    for (int i = tid; i < D_DICT; i += 256)
        buf[i] = __builtin_nontemporal_load(&src[i]);
    __syncthreads();

    for (int s = 0; s < KSEL; ++s) {
        float m = -1.0f;
        int loc = 0;
        for (int i = tid; i < D_DICT; i += 256) {
            float v = buf[i];
            if (v > m) { m = v; loc = i; }
        }
        // wave32 reduction
#pragma unroll
        for (int off = 16; off > 0; off >>= 1) {
            float om = __shfl_xor(m, off, 32);
            int   ol = __shfl_xor(loc, off, 32);
            if (om > m) { m = om; loc = ol; }
        }
        if (lane == 0) { wmaxs[wave] = m; wlocs[wave] = loc; }
        __syncthreads();
        if (tid == 0) {
            float bm = wmaxs[0];
            int   bl = wlocs[0];
#pragma unroll
            for (int w = 1; w < 8; ++w)
                if (wmaxs[w] > bm) { bm = wmaxs[w]; bl = wlocs[w]; }
            vals[(size_t)row * KSEL + s] = bm;
            idxs[(size_t)row * KSEL + s] = bl;
            buf[bl] = -__builtin_inff();   // remove from further consideration
        }
        __syncthreads();
    }
}

// ---------------------------------------------------------------------------
// Sparse decoder: x_hat[b, :] = sum_j vals[b,j] * W_enc[idx[b,j], :] + b_dec
// (W_dec column f == W_enc row f, which is contiguous -> coalesced gathers,
//  and W_enc stays hot in the 192MB L2.)
// One workgroup per row; 256 threads x 3 output dims each (768 = 3*256).
// ---------------------------------------------------------------------------
__global__ __launch_bounds__(256) void dec_kernel(const float* __restrict__ Wenc,
                                                  const float* __restrict__ vals,
                                                  const int* __restrict__ idxs,
                                                  const float* __restrict__ b_dec,
                                                  float* __restrict__ out) {
    __shared__ float sv[KSEL];
    __shared__ int   si[KSEL];

    const int row = blockIdx.x;
    const int tid = threadIdx.x;
    if (tid < KSEL) {
        sv[tid] = vals[(size_t)row * KSEL + tid];
        si[tid] = idxs[(size_t)row * KSEL + tid];
    }
    __syncthreads();

    float acc0 = b_dec[tid];
    float acc1 = b_dec[tid + 256];
    float acc2 = b_dec[tid + 512];

#pragma unroll 4
    for (int j = 0; j < KSEL; ++j) {
        const float v = sv[j];
        const float* wr = Wenc + (size_t)si[j] * D_ACT;
        acc0 += v * wr[tid];
        acc1 += v * wr[tid + 256];
        acc2 += v * wr[tid + 512];
    }

    float* o = out + (size_t)row * D_ACT;
    o[tid]       = acc0;
    o[tid + 256] = acc1;
    o[tid + 512] = acc2;
}

// ---------------------------------------------------------------------------
extern "C" void kernel_launch(void* const* d_in, const int* in_sizes, int n_in,
                              void* d_out, int out_size, void* d_ws, size_t ws_size,
                              hipStream_t stream) {
    const float* x     = (const float*)d_in[0];   // [4096, 768]
    const float* W_enc = (const float*)d_in[1];   // [24576, 768]
    const float* b_enc = (const float*)d_in[2];   // [24576]
    // d_in[3] = W_dec [768, 24576] (tied: column f == W_enc row f; unused)
    const float* b_dec = (const float*)d_in[4];   // [768]
    float* x_hat = (float*)d_out;                 // [4096, 768]

    char* ws = (char*)d_ws;
    size_t off = 0;
    __bf16* x_bf = (__bf16*)(ws + off); off += (size_t)BATCH * D_ACT * sizeof(__bf16);
    __bf16* w_bf = (__bf16*)(ws + off); off += (size_t)D_DICT * D_ACT * sizeof(__bf16);
    float*  mu   = (float*)(ws + off);  off += (size_t)BATCH * D_DICT * sizeof(float);
    float*  vals = (float*)(ws + off);  off += (size_t)BATCH * KSEL * sizeof(float);
    int*    idxs = (int*)(ws + off);    off += (size_t)BATCH * KSEL * sizeof(int);

    // 1) fp32 -> bf16 for GEMM operands
    {
        long n4 = (long)BATCH * D_ACT / 4;
        cvt_bf16_kernel<<<(n4 + 255) / 256, 256, 0, stream>>>(x, x_bf, n4);
    }
    {
        long n4 = (long)D_DICT * D_ACT / 4;
        cvt_bf16_kernel<<<(n4 + 255) / 256, 256, 0, stream>>>(W_enc, w_bf, n4);
    }

    // 2) encoder GEMM + bias + relu (WMMA bf16, f32 accumulate)
    {
        dim3 grid(BATCH / 128, D_DICT / 256);   // 32 x 96
        enc_wmma_kernel<<<grid, 256, 0, stream>>>(x_bf, w_bf, b_enc, mu);
    }

    // 3) top-k selection per row
    topk_kernel<<<BATCH, 256, 0, stream>>>(mu, vals, idxs);

    // 4) sparse decoder
    dec_kernel<<<BATCH, 256, 0, stream>>>(W_enc, vals, idxs, b_dec, x_hat);
}
